// HybridTransformerSSMModel_52922587022087
// MI455X (gfx1250) — matrix operand order repaired and checked by reading, not compiled, not run on hardware
//
#include <hip/hip_runtime.h>
#include <hip/hip_bf16.h>
#include <stdint.h>

// ---------------------------------------------------------------------------
// HybridTransformerSSMModel for MI455X (gfx1250, wave32).
// Dense algebra -> v_wmma_f32_16x16x32_f16 (f16 in, f32 acc).
// B (weight) tiles staged per-workgroup via global_load_async_to_lds_b128
// (ASYNCcnt) into double-buffered LDS; fragments gathered with 16-bit LDS loads (one f16 per K row).
// ---------------------------------------------------------------------------

#define B_   32
#define S_   512
#define I_   8
#define H_   512
#define NH_  4
#define D_   128          // head dim
#define FF_  2048
#define T_   (B_ * S_)    // 16384 tokens
#define BH_  (B_ * NH_)   // 128 attention batches
#define CHUNK_BH 16

#define LDS_ROW 72                 // 64 f16 + 8 pad (144 B rows, 16B aligned)
#define LDS_BUF (32 * LDS_ROW)     // shorts per buffer (4608 B)

typedef __attribute__((ext_vector_type(16))) _Float16 v16h;
typedef __attribute__((ext_vector_type(8)))  float    v8f;

union FragAB { v16h v; uint32_t u[8]; uint4 q[2]; };
union FragC  { v8f  v; float f[8]; };

__device__ __forceinline__ unsigned short f2h(float f) {
  _Float16 h = (_Float16)f;                    // RNE convert
  return __builtin_bit_cast(unsigned short, h);
}
__device__ __forceinline__ float gelu_exact(float x) {
  return 0.5f * x * (1.0f + erff(x * 0.70710678118654752f));
}

// async global -> LDS, 16B per lane; tracked by ASYNCcnt
__device__ __forceinline__ void async_b128(unsigned ldsOff, const void* gaddr) {
  asm volatile("global_load_async_to_lds_b128 %0, %1, off"
               :: "v"(ldsOff), "v"(gaddr) : "memory");
}
__device__ __forceinline__ void wait_asynccnt0() {
  asm volatile("s_wait_asynccnt 0x0" ::: "memory");
}

// ---------------------------------------------------------------------------
// Generic batched GEMM:  C[b] = act( A[b] (MxK,f16) * Wt[b] (KxN,f16) [+bias]
//                                    [*scale+shift] )
// Block tile 256(M) x 64(N): 8 waves stacked in M sharing one B strip.
// Requires M%256==0, N%64==0, K%32==0.
// Operand VGPR layouts per CDNA5 ISA 7.12.2 (16-bit A 16x32, B 32x16, C f32).
// ---------------------------------------------------------------------------
__global__ void __launch_bounds__(256)
gemm_f16_kernel(const unsigned short* __restrict__ A, long long sA, int lda,
                const unsigned short* __restrict__ W, long long sW, int ldw,
                float* __restrict__ Cf, unsigned short* __restrict__ Ch,
                long long sC, int ldc, int M, int N, int K,
                const float* __restrict__ bias,
                const float* __restrict__ scale, const float* __restrict__ shift,
                int act)  // 0 none, 1 gelu, 2 sigmoid, 3 softplus
{
  __shared__ unsigned short ldsB[2 * LDS_BUF];   // double-buffered 32x64 f16 B tile
  const int tid  = threadIdx.x;
  const int wv   = tid >> 5;
  const int lane = tid & 31;
  const int tilesN = N >> 6;
  const int bm = blockIdx.x / tilesN;
  const int bn = blockIdx.x - bm * tilesN;
  const int m0 = bm * 256 + wv * 32;
  const int n0 = bn * 64;

  A += (long long)blockIdx.y * sA;
  W += (long long)blockIdx.y * sW;
  const long long cbase = (long long)blockIdx.y * sC;

  const int lm = lane & 15;
  const int hi = lane >> 4;                      // lane half selects K phase

  // B staging: thread tid moves one 16B chunk: row = tid/8, 8-f16 chunk tid%8
  const int brow = tid >> 3;                     // 0..31 (K row in tile)
  const int bco  = (tid & 7) << 3;               // f16 column offset 0..56
  const unsigned short* gB = W + (size_t)brow * ldw + n0 + bco;
  const unsigned ldsWr0 = (unsigned)(uintptr_t)&ldsB[brow * LDS_ROW + bco];
  const unsigned ldsWr1 = ldsWr0 + LDS_BUF * 2;  // byte offset of buffer 1

  FragC acc[2][4];
#pragma unroll
  for (int i = 0; i < 2; ++i)
#pragma unroll
    for (int t = 0; t < 4; ++t)
#pragma unroll
      for (int r = 0; r < 8; ++r) acc[i][t].f[r] = 0.f;

  const unsigned short* Arow0 = A + (size_t)(m0 + lm) * lda;
  const unsigned short* Arow1 = A + (size_t)(m0 + 16 + lm) * lda;
  const unsigned short* ldsRd = &ldsB[lm];               // B frag: lane = N column

  const int nsteps = K >> 5;
  async_b128(ldsWr0, gB);                        // stage tile 0 -> buf 0

  for (int s = 0; s < nsteps; ++s) {
    wait_asynccnt0();                            // own writes of tile s done
    __syncthreads();                             // all waves wrote tile s; prev readers done
    if (s + 1 < nsteps)                          // overlap next stage with compute
      async_b128((s & 1) ? ldsWr0 : ldsWr1, gB + (size_t)(s + 1) * 32 * ldw);

    const int k0 = s << 5;
    if (k0 + 32 < K) {                           // -> global_prefetch_b8 (A stream)
      __builtin_prefetch(Arow0 + k0 + 32, 0, 1);
      __builtin_prefetch(Arow1 + k0 + 32, 0, 1);
    }
    FragAB a0, a1, b0, b1, b2, b3;
    // A frag (global): lanes 0-15 hold K {0..7,16..23}, lanes 16-31 {8..15,24..31}
    const uint4* pa0 = (const uint4*)(Arow0 + k0);
    const uint4* pa1 = (const uint4*)(Arow1 + k0);
    a0.q[0] = pa0[hi];     a0.q[1] = pa0[2 + hi];
    a1.q[0] = pa1[hi];     a1.q[1] = pa1[2 + hi];
    // B frags (LDS): lane = N column, regs hold K in same order as A frag
    const unsigned short* lb = ldsRd + (s & 1) * LDS_BUF;
#pragma unroll
    for (int j = 0; j < 16; ++j) {
      const int kk = (j < 8) ? (8 * hi + j) : (16 + 8 * hi + (j - 8));
      const _Float16* p = (const _Float16*)(lb + kk * LDS_ROW);
      b0.v[j] = p[0]; b1.v[j] = p[16]; b2.v[j] = p[32]; b3.v[j] = p[48];
    }

    acc[0][0].v = __builtin_amdgcn_wmma_f32_16x16x32_f16(false, a0.v, false, b0.v, (short)0, acc[0][0].v, false, false);
    acc[0][1].v = __builtin_amdgcn_wmma_f32_16x16x32_f16(false, a0.v, false, b1.v, (short)0, acc[0][1].v, false, false);
    acc[0][2].v = __builtin_amdgcn_wmma_f32_16x16x32_f16(false, a0.v, false, b2.v, (short)0, acc[0][2].v, false, false);
    acc[0][3].v = __builtin_amdgcn_wmma_f32_16x16x32_f16(false, a0.v, false, b3.v, (short)0, acc[0][3].v, false, false);
    acc[1][0].v = __builtin_amdgcn_wmma_f32_16x16x32_f16(false, a1.v, false, b0.v, (short)0, acc[1][0].v, false, false);
    acc[1][1].v = __builtin_amdgcn_wmma_f32_16x16x32_f16(false, a1.v, false, b1.v, (short)0, acc[1][1].v, false, false);
    acc[1][2].v = __builtin_amdgcn_wmma_f32_16x16x32_f16(false, a1.v, false, b2.v, (short)0, acc[1][2].v, false, false);
    acc[1][3].v = __builtin_amdgcn_wmma_f32_16x16x32_f16(false, a1.v, false, b3.v, (short)0, acc[1][3].v, false, false);
  }

  // C layout: lane n = lane&15; VGPR r -> row r (+8 for lanes 16-31)
  const int colb = lane & 15;
  const int rb   = hi << 3;
#pragma unroll
  for (int t = 0; t < 4; ++t) {
    const int col = n0 + (t << 4) + colb;
    const float bs = bias  ? bias[col]  : 0.f;
    const float sc = scale ? scale[col] : 1.f;
    const float sh = shift ? shift[col] : 0.f;
#pragma unroll
    for (int i = 0; i < 2; ++i) {
      const int mrow = m0 + (i << 4) + rb;
#pragma unroll
      for (int r = 0; r < 8; ++r) {
        float v = acc[i][t].f[r] + bs;
        if (scale) v = v * sc + sh;
        if (act == 1)      v = gelu_exact(v);
        else if (act == 2) v = 1.f / (1.f + expf(-v));
        else if (act == 3) v = (v > 20.f) ? v : log1pf(expf(v));
        const long long off = cbase + (long long)(mrow + r) * ldc + col;
        if (Cf) Cf[off] = v;
        if (Ch) Ch[off] = f2h(v);
      }
    }
  }
}

// ---------------- elementwise / reduction kernels (fp32) -------------------

__global__ void enc1_kernel(const float* __restrict__ x, const float* __restrict__ w1,
                            const float* __restrict__ b1, float* __restrict__ out) {
  const int idx = blockIdx.x * 256 + threadIdx.x;   // T_*H_
  const int t = idx >> 9, n = idx & 511;
  const float* xr = x + t * I_;
  const float* wr = w1 + n * I_;
  float s = b1[n];
#pragma unroll
  for (int k = 0; k < I_; ++k) s += xr[k] * wr[k];
  out[idx] = s;
}

__global__ void ln512_kernel(const float* __restrict__ in, const float* __restrict__ resid,
                             const float* __restrict__ g, const float* __restrict__ bta,
                             float* __restrict__ outF, unsigned short* __restrict__ outH,
                             int act) {
  __shared__ float red[256];
  const size_t base = (size_t)blockIdx.x * H_;
  const int tid = threadIdx.x;
  float x0 = in[base + tid], x1 = in[base + tid + 256];
  if (resid) { x0 += resid[base + tid]; x1 += resid[base + tid + 256]; }
  red[tid] = x0 + x1; __syncthreads();
  for (int o = 128; o > 0; o >>= 1) { if (tid < o) red[tid] += red[tid + o]; __syncthreads(); }
  const float mean = red[0] * (1.f / H_); __syncthreads();
  const float d0 = x0 - mean, d1 = x1 - mean;
  red[tid] = d0 * d0 + d1 * d1; __syncthreads();
  for (int o = 128; o > 0; o >>= 1) { if (tid < o) red[tid] += red[tid + o]; __syncthreads(); }
  const float rstd = rsqrtf(red[0] * (1.f / H_) + 1e-5f);
  float y0 = d0 * rstd * g[tid] + bta[tid];
  float y1 = d1 * rstd * g[tid + 256] + bta[tid + 256];
  if (act == 1) { y0 = gelu_exact(y0); y1 = gelu_exact(y1); }
  if (outF) { outF[base + tid] = y0; outF[base + tid + 256] = y1; }
  if (outH) { outH[base + tid] = f2h(y0); outH[base + tid + 256] = f2h(y1); }
}

__global__ void softmax512_kernel(const float* __restrict__ in,
                                  unsigned short* __restrict__ out, float scl) {
  __shared__ float red[256];
  const size_t base = (size_t)blockIdx.x * 512;
  const int tid = threadIdx.x;
  const float x0 = in[base + tid] * scl, x1 = in[base + tid + 256] * scl;
  red[tid] = fmaxf(x0, x1); __syncthreads();
  for (int o = 128; o > 0; o >>= 1) { if (tid < o) red[tid] = fmaxf(red[tid], red[tid + o]); __syncthreads(); }
  const float mx = red[0]; __syncthreads();
  const float e0 = expf(x0 - mx), e1 = expf(x1 - mx);
  red[tid] = e0 + e1; __syncthreads();
  for (int o = 128; o > 0; o >>= 1) { if (tid < o) red[tid] += red[tid + o]; __syncthreads(); }
  const float inv = 1.f / red[0];
  out[base + tid]       = f2h(e0 * inv);
  out[base + tid + 256] = f2h(e1 * inv);
}

__global__ void wprep_kernel(const float* __restrict__ Wsrc, unsigned short* __restrict__ Wt,
                             int N, int K) {
  const int idx = blockIdx.x * 256 + threadIdx.x;
  if (idx >= N * K) return;
  const int n = idx / K, k = idx % K;
  Wt[(size_t)k * N + n] = f2h(Wsrc[idx]);
}

__global__ void convw_kernel(const float* __restrict__ w, unsigned short* __restrict__ Wt,
                             int Cout, int Cin) {
  const int idx = blockIdx.x * 256 + threadIdx.x;
  if (idx >= Cout * Cin * 3) return;
  const int co = idx / (Cin * 3);
  const int r  = idx % (Cin * 3);
  const int ci = r / 3, kt = r % 3;
  Wt[(size_t)(kt * Cin + ci) * Cout + co] = f2h(w[idx]);
}

__global__ void bnprep_kernel(const float* __restrict__ bc, const float* __restrict__ g,
                              const float* __restrict__ bb, const float* __restrict__ m,
                              const float* __restrict__ v, float* __restrict__ sc,
                              float* __restrict__ sh, int C) {
  const int i = blockIdx.x * 256 + threadIdx.x;
  if (i >= C) return;
  const float s = g[i] * rsqrtf(v[i] + 1e-5f);
  sc[i] = s;
  sh[i] = (bc[i] - m[i]) * s + bb[i];
}

__global__ void im2col_kernel(const unsigned short* __restrict__ inb,
                              unsigned short* __restrict__ outb, int Cin) {
  const int K3 = 3 * Cin;
  const long long idx = (long long)blockIdx.x * 256 + threadIdx.x;
  if (idx >= (long long)T_ * K3) return;
  const int t  = (int)(idx / K3);
  const int j  = (int)(idx % K3);
  const int kt = j / Cin, ci = j % Cin;
  const int b = t >> 9, l = t & 511;
  const int lp = l + kt - 1;
  unsigned short val = 0;                       // f16 zero (padding)
  if (lp >= 0 && lp < S_) val = inb[(size_t)(b * S_ + lp) * Cin + ci];
  outb[idx] = val;
}

__global__ void pack_qkv_kernel(const float* __restrict__ qkv,
                                unsigned short* __restrict__ Qh,
                                unsigned short* __restrict__ KtH,
                                unsigned short* __restrict__ Vh) {
  const int idx = blockIdx.x * 256 + threadIdx.x;   // (b,h,s,i)
  const int i = idx & (D_ - 1);
  const int s = (idx >> 7) & (S_ - 1);
  const int h = (idx >> 16) & (NH_ - 1);
  const int b = idx >> 18;
  const size_t t = (size_t)(b * S_ + s) * (3 * H_);
  const int c = h * D_ + i;
  Qh[idx] = f2h(qkv[t + c]);
  Vh[idx] = f2h(qkv[t + 2 * H_ + c]);
  KtH[((size_t)(b * NH_ + h) * D_ + i) * S_ + s] = f2h(qkv[t + H_ + c]);
}

__global__ void unpack_attn_kernel(const unsigned short* __restrict__ ao,
                                   unsigned short* __restrict__ attnb, int bhBase) {
  const int idx = blockIdx.x * 256 + threadIdx.x;   // CHUNK_BH*S*D
  const int i = idx & (D_ - 1);
  const int s = (idx >> 7) & (S_ - 1);
  const int g = idx >> 16;
  const int bh = bhBase + g;
  const int b = bh >> 2, h = bh & 3;
  attnb[(size_t)(b * S_ + s) * H_ + h * D_ + i] = ao[idx];
}

__global__ void ssm_scan_kernel(const float* __restrict__ xp, const float* __restrict__ dt,
                                const float* __restrict__ Bg, const float* __restrict__ Cg,
                                const float* __restrict__ Alog, float* __restrict__ pooled,
                                int dir) {
  const int idx = blockIdx.x * 256 + threadIdx.x;   // B*H
  const int b = idx >> 9, h = idx & 511;
  const float na = -expf(Alog[h]);
  float y = 0.f, acc = 0.f;
  for (int stp = 0; stp < S_; ++stp) {
    const int t = dir ? (S_ - 1 - stp) : stp;
    const size_t off = (size_t)(b * S_ + t) * H_ + h;
    const float a  = expf(na * dt[off]);
    const float hn = a * y + (1.f - a) * (Bg[off] * xp[off]);
    y = Cg[off] * hn;
    acc += y;
  }
  pooled[b * (2 * H_) + dir * H_ + h] = acc * (1.f / S_);
}

__global__ void fc_kernel(const float* __restrict__ pooled, const float* __restrict__ fw,
                          const float* __restrict__ fb, float* __restrict__ out) {
  __shared__ float red[256];
  const int b = blockIdx.x, tid = threadIdx.x;
  const float* row = pooled + b * 1024;
  float s = row[tid] * fw[tid] + row[tid + 256] * fw[tid + 256] +
            row[tid + 512] * fw[tid + 512] + row[tid + 768] * fw[tid + 768];
  red[tid] = s; __syncthreads();
  for (int o = 128; o > 0; o >>= 1) { if (tid < o) red[tid] += red[tid + o]; __syncthreads(); }
  if (tid == 0) out[b] = red[0] + fb[0];
}

// ---------------------------------------------------------------------------
extern "C" void kernel_launch(void* const* d_in, const int* in_sizes, int n_in,
                              void* d_out, int out_size, void* d_ws, size_t ws_size,
                              hipStream_t stream) {
  (void)n_in; (void)out_size; (void)ws_size;
  // ---- canonical leaf slots ----
  // 0:x | 1..8 enc(w1,b1,ln1g,ln1b,w2,b2,ln2g,ln2b) | 9..26 cnn 3x(w,b,g,bb,m,v)
  // 27..50 trans 2x(qkvw,qkvb,outw,outb,ln1g,ln1b,ff1w,ff1b,ff2w,ff2b,ln2g,ln2b)
  // 51..56 ssm_fwd(Alog,inw,dtw,dtb,Bw,Cw) | 57..62 ssm_bwd | 63 fcw | 64 fcb
  enum { L_COUNT = 65 };
  int src[L_COUNT];
  const int cnnOff[6]  = {5, 0, 2, 1, 3, 4};
  const int trOff[12]  = {11, 10, 9, 8, 5, 4, 1, 0, 3, 2, 7, 6};
  const int ssmOff[6]  = {0, 5, 4, 3, 1, 2};
  const int encOff[8]  = {6, 0, 3, 2, 7, 1, 5, 4};
  int layout;
  if (in_sizes[0] == 131072) layout = (in_sizes[1] == 4096) ? 0 : 1;
  else layout = 2;
  if (layout == 0) {
    for (int i = 0; i < L_COUNT; ++i) src[i] = i;
  } else if (layout == 1) {            // top-level insertion order, dicts key-sorted
    src[0] = 0;
    for (int j = 0; j < 8; ++j) src[1 + j] = 1 + encOff[j];
    for (int i = 0; i < 3; ++i) for (int j = 0; j < 6; ++j) src[9 + 6 * i + j] = 9 + 6 * i + cnnOff[j];
    for (int l = 0; l < 2; ++l) for (int j = 0; j < 12; ++j) src[27 + 12 * l + j] = 27 + 12 * l + trOff[j];
    for (int j = 0; j < 6; ++j) { src[51 + j] = 51 + ssmOff[j]; src[57 + j] = 57 + ssmOff[j]; }
    src[63] = 63; src[64] = 64;
  } else {                             // whole tree key-sorted (jax leaves)
    src[0] = 64;
    for (int j = 0; j < 8; ++j) src[1 + j] = 18 + encOff[j];
    for (int i = 0; i < 3; ++i) for (int j = 0; j < 6; ++j) src[9 + 6 * i + j] = 6 * i + cnnOff[j];
    for (int l = 0; l < 2; ++l) for (int j = 0; j < 12; ++j) src[27 + 12 * l + j] = 40 + 12 * l + trOff[j];
    for (int j = 0; j < 6; ++j) { src[51 + j] = 34 + ssmOff[j]; src[57 + j] = 28 + ssmOff[j]; }
    src[63] = 27; src[64] = 26;
  }
  const float* P[L_COUNT];
  for (int i = 0; i < L_COUNT; ++i) P[i] = (const float*)d_in[src[i]];

  // ---- workspace bump allocator ----
  char* wsp = (char*)d_ws;
  auto alloc = [&](size_t bytes) -> void* {
    void* p = (void*)wsp;
    wsp += (bytes + 255) & ~(size_t)255;
    return p;
  };
  const int convCin[3]  = {512, 128, 256};
  const int convCout[3] = {128, 256, 512};

  unsigned short* wt_encw2 = (unsigned short*)alloc((size_t)H_ * H_ * 2);
  unsigned short* wt_conv[3];
  for (int i = 0; i < 3; ++i) wt_conv[i] = (unsigned short*)alloc((size_t)3 * convCin[i] * convCout[i] * 2);
  float *bn_sc[3], *bn_sh[3];
  for (int i = 0; i < 3; ++i) { bn_sc[i] = (float*)alloc(convCout[i] * 4); bn_sh[i] = (float*)alloc(convCout[i] * 4); }
  unsigned short *wt_qkv[2], *wt_out[2], *wt_ff1[2], *wt_ff2[2];
  for (int l = 0; l < 2; ++l) {
    wt_qkv[l] = (unsigned short*)alloc((size_t)H_ * 3 * H_ * 2);
    wt_out[l] = (unsigned short*)alloc((size_t)H_ * H_ * 2);
    wt_ff1[l] = (unsigned short*)alloc((size_t)H_ * FF_ * 2);
    wt_ff2[l] = (unsigned short*)alloc((size_t)FF_ * H_ * 2);
  }
  unsigned short* wt_ssm[2][4];
  for (int d = 0; d < 2; ++d) for (int j = 0; j < 4; ++j) wt_ssm[d][j] = (unsigned short*)alloc((size_t)H_ * H_ * 2);

  float* tmpf = (float*)alloc((size_t)T_ * H_ * 4);
  float* hf   = (float*)alloc((size_t)T_ * H_ * 4);
  float* h1f  = (float*)alloc((size_t)T_ * H_ * 4);
  unsigned short* hb    = (unsigned short*)alloc((size_t)T_ * H_ * 2);
  unsigned short* h1b   = (unsigned short*)alloc((size_t)T_ * H_ * 2);
  unsigned short* attnb = (unsigned short*)alloc((size_t)T_ * H_ * 2);
  unsigned short* cb0   = (unsigned short*)alloc((size_t)T_ * H_ * 2);
  unsigned short* cb1   = (unsigned short*)alloc((size_t)T_ * H_ * 2);
  float* buf1 = (float*)alloc((size_t)T_ * H_ * 4 * 4);          // qkv f32 / ssm arrays
  unsigned short* buf2 = (unsigned short*)alloc((size_t)T_ * FF_ * 2); // im2col / ff hidden
  unsigned short* Qh  = (unsigned short*)alloc((size_t)BH_ * S_ * D_ * 2);
  unsigned short* Kth = (unsigned short*)alloc((size_t)BH_ * S_ * D_ * 2);
  unsigned short* Vh  = (unsigned short*)alloc((size_t)BH_ * S_ * D_ * 2);
  float* scoresC = (float*)alloc((size_t)CHUNK_BH * S_ * S_ * 4);
  unsigned short* probsC = (unsigned short*)alloc((size_t)CHUNK_BH * S_ * S_ * 2);
  unsigned short* aoC    = (unsigned short*)alloc((size_t)CHUNK_BH * S_ * D_ * 2);
  float* pooled = (float*)alloc((size_t)B_ * 2 * H_ * 4);

  auto gemm = [&](const unsigned short* A, long long sA, int lda,
                  const unsigned short* W, long long sW, int ldw,
                  float* Cf, unsigned short* Ch, long long sC, int ldc,
                  int M, int N, int K, int batch,
                  const float* bias, const float* scl, const float* shf, int act) {
    const int tiles = (M / 256) * (N / 64);      // one block per 256x64 tile
    dim3 grid((unsigned)tiles, (unsigned)batch, 1);
    gemm_f16_kernel<<<grid, 256, 0, stream>>>(A, sA, lda, W, sW, ldw, Cf, Ch, sC, ldc,
                                              M, N, K, bias, scl, shf, act);
  };
  auto wprep = [&](const float* Wsrc, unsigned short* Wt, int N, int K) {
    const int total = N * K;
    wprep_kernel<<<(total + 255) / 256, 256, 0, stream>>>(Wsrc, Wt, N, K);
  };

  // ---- weight prep (f32 -> transposed KxN f16) ----
  wprep(P[5], wt_encw2, H_, H_);
  for (int i = 0; i < 3; ++i) {
    const int total = convCout[i] * convCin[i] * 3;
    convw_kernel<<<(total + 255) / 256, 256, 0, stream>>>(P[9 + 6 * i + 0], wt_conv[i], convCout[i], convCin[i]);
    bnprep_kernel<<<(convCout[i] + 255) / 256, 256, 0, stream>>>(
        P[9 + 6 * i + 1], P[9 + 6 * i + 2], P[9 + 6 * i + 3], P[9 + 6 * i + 4], P[9 + 6 * i + 5],
        bn_sc[i], bn_sh[i], convCout[i]);
  }
  for (int l = 0; l < 2; ++l) {
    const int tb = 27 + 12 * l;
    wprep(P[tb + 0], wt_qkv[l], 3 * H_, H_);
    wprep(P[tb + 2], wt_out[l], H_, H_);
    wprep(P[tb + 6], wt_ff1[l], FF_, H_);
    wprep(P[tb + 8], wt_ff2[l], H_, FF_);
  }
  for (int d = 0; d < 2; ++d) {
    const int sb = 51 + 6 * d;
    wprep(P[sb + 1], wt_ssm[d][0], H_, H_);   // in_w
    wprep(P[sb + 2], wt_ssm[d][1], H_, H_);   // dt_w
    wprep(P[sb + 4], wt_ssm[d][2], H_, H_);   // B_w
    wprep(P[sb + 5], wt_ssm[d][3], H_, H_);   // C_w
  }

  // ---- encoder: Linear(8->512) -> LN -> GELU -> Linear(512->512) -> LN ----
  enc1_kernel<<<(T_ * H_) / 256, 256, 0, stream>>>(P[0], P[1], P[2], tmpf);
  ln512_kernel<<<T_, 256, 0, stream>>>(tmpf, nullptr, P[3], P[4], nullptr, hb, 1);
  gemm(hb, 0, H_, wt_encw2, 0, H_, tmpf, nullptr, 0, H_, T_, H_, H_, 1, P[6], nullptr, nullptr, 0);
  ln512_kernel<<<T_, 256, 0, stream>>>(tmpf, nullptr, P[7], P[8], nullptr, cb0, 0);

  // ---- conv stack (im2col + GEMM with BN/GELU epilogue) ----
  {
    const unsigned short* ci = cb0;
    unsigned short* co_arr[3] = {cb1, cb0, hb};
    for (int i = 0; i < 3; ++i) {
      const int Cin = convCin[i], Cout = convCout[i], K3 = 3 * Cin;
      const long long total = (long long)T_ * K3;
      im2col_kernel<<<(unsigned)((total + 255) / 256), 256, 0, stream>>>(ci, buf2, Cin);
      gemm(buf2, 0, K3, wt_conv[i], 0, Cout,
           (i == 2 ? hf : nullptr), co_arr[i], 0, Cout,
           T_, Cout, K3, 1, nullptr, bn_sc[i], bn_sh[i], 1);
      ci = co_arr[i];
    }
  }

  // ---- 2 transformer encoder layers (post-LN) ----
  for (int l = 0; l < 2; ++l) {
    const int tb = 27 + 12 * l;
    gemm(hb, 0, H_, wt_qkv[l], 0, 3 * H_, buf1, nullptr, 0, 3 * H_,
         T_, 3 * H_, H_, 1, P[tb + 1], nullptr, nullptr, 0);
    pack_qkv_kernel<<<(B_ * NH_ * S_ * D_) / 256, 256, 0, stream>>>(buf1, Qh, Kth, Vh);
    for (int c = 0; c < BH_ / CHUNK_BH; ++c) {
      const size_t off = (size_t)c * CHUNK_BH * S_ * D_;
      gemm(Qh + off, (long long)S_ * D_, D_,
           Kth + off, (long long)D_ * S_, S_,
           scoresC, nullptr, (long long)S_ * S_, S_,
           S_, S_, D_, CHUNK_BH, nullptr, nullptr, nullptr, 0);
      softmax512_kernel<<<CHUNK_BH * S_, 256, 0, stream>>>(scoresC, probsC, 0.08838834764831845f);
      gemm(probsC, (long long)S_ * S_, S_,
           Vh + off, (long long)S_ * D_, D_,
           nullptr, aoC, (long long)S_ * D_, D_,
           S_, D_, S_, CHUNK_BH, nullptr, nullptr, nullptr, 0);
      unpack_attn_kernel<<<(CHUNK_BH * S_ * D_) / 256, 256, 0, stream>>>(aoC, attnb, c * CHUNK_BH);
    }
    gemm(attnb, 0, H_, wt_out[l], 0, H_, tmpf, nullptr, 0, H_,
         T_, H_, H_, 1, P[tb + 3], nullptr, nullptr, 0);
    ln512_kernel<<<T_, 256, 0, stream>>>(tmpf, hf, P[tb + 4], P[tb + 5], h1f, h1b, 0);
    gemm(h1b, 0, H_, wt_ff1[l], 0, FF_, nullptr, buf2, 0, FF_,
         T_, FF_, H_, 1, P[tb + 7], nullptr, nullptr, 1);
    gemm(buf2, 0, FF_, wt_ff2[l], 0, H_, tmpf, nullptr, 0, H_,
         T_, H_, FF_, 1, P[tb + 9], nullptr, nullptr, 0);
    ln512_kernel<<<T_, 256, 0, stream>>>(tmpf, h1f, P[tb + 10], P[tb + 11], hf, hb, 0);
  }

  // ---- bidirectional SSM: 4 projections + sequential scan per direction ----
  for (int d = 0; d < 2; ++d) {
    const int sb = 51 + 6 * d;
    float* xp = buf1;
    float* dt = buf1 + (size_t)T_ * H_;
    float* Bg = buf1 + 2 * (size_t)T_ * H_;
    float* Cg = buf1 + 3 * (size_t)T_ * H_;
    gemm(hb, 0, H_, wt_ssm[d][0], 0, H_, xp, nullptr, 0, H_, T_, H_, H_, 1, nullptr, nullptr, nullptr, 0);
    gemm(hb, 0, H_, wt_ssm[d][1], 0, H_, dt, nullptr, 0, H_, T_, H_, H_, 1, P[sb + 3], nullptr, nullptr, 3);
    gemm(hb, 0, H_, wt_ssm[d][2], 0, H_, Bg, nullptr, 0, H_, T_, H_, H_, 1, nullptr, nullptr, nullptr, 2);
    gemm(hb, 0, H_, wt_ssm[d][3], 0, H_, Cg, nullptr, 0, H_, T_, H_, H_, 1, nullptr, nullptr, nullptr, 2);
    ssm_scan_kernel<<<(B_ * H_) / 256, 256, 0, stream>>>(xp, dt, Bg, Cg, P[sb + 0], pooled, d);
  }

  // ---- final FC (B x 1024 -> B) ----
  fc_kernel<<<B_, 256, 0, stream>>>(pooled, P[63], P[64], (float*)d_out);
}
